// SPTransformerBlock_89215060672750
// MI455X (gfx1250) — matrix operand order repaired and checked by reading, not compiled, not run on hardware
//
#include <hip/hip_runtime.h>
#include <hip/hip_bf16.h>

// MI455X / gfx1250 transformer block.
// - All weights converted fp32->f16 once (~134MB, ~fits the 192MB L2);
//   every GEMM / attention matmul runs on v_wmma_f32_16x16x32_f16.
// - GEMMs are software-pipelined double-buffered: A tiles stream directly
//   global->LDS via global_load_async_to_lds_b128 (ASYNCcnt), B tiles are
//   prefetched into VGPRs one K-step ahead and transposed into LDS.
// - Flash attention: online softmax, K tiles via async-to-LDS, P does a
//   wave-local LDS round trip guarded by s_wait_dscnt.

typedef __attribute__((ext_vector_type(16))) _Float16 v16h;
typedef __attribute__((ext_vector_type(8)))  _Float16 v8h;
typedef __attribute__((ext_vector_type(8)))  float    v8f;

static __device__ __forceinline__ v8f wmma_f16(v16h a, v16h b, v8f c) {
  return __builtin_amdgcn_wmma_f32_16x16x32_f16(false, a, false, b, (short)0, c,
                                                false, false);
}

static __device__ __forceinline__ v16h cat8(v8h lo, v8h hi) {
  v16h r;
#pragma unroll
  for (int i = 0; i < 8; ++i) { r[i] = lo[i]; r[i + 8] = hi[i]; }
  return r;
}

static __device__ __forceinline__ float halfmax16(float v) {
#pragma unroll
  for (int off = 1; off < 16; off <<= 1) v = fmaxf(v, __shfl_xor(v, off, 32));
  return v;
}
static __device__ __forceinline__ float halfsum16(float v) {
#pragma unroll
  for (int off = 1; off < 16; off <<= 1) v += __shfl_xor(v, off, 32);
  return v;
}

// Async DMA: 16 bytes global -> LDS, tracked by ASYNCcnt (gfx1250 path).
// Generic shared pointers: addr[31:0] IS the LDS byte address (aperture rule).
static __device__ __forceinline__ void async_load_b128(const void* g, void* l) {
  unsigned lds = (unsigned)(unsigned long long)l;
  asm volatile("global_load_async_to_lds_b128 %0, %1, off"
               :: "v"(lds), "v"(g) : "memory");
}
static __device__ __forceinline__ void wait_async0() {
  asm volatile("s_wait_asynccnt 0x0" ::: "memory");
}
static __device__ __forceinline__ void wait_ds0() {
  asm volatile("s_wait_dscnt 0x0" ::: "memory");
}

// ---------------------------------------------------------------------------
// fp32 -> f16 weight conversion (one HBM pass; thereafter weights live in L2)
// ---------------------------------------------------------------------------
__global__ void conv_f32_f16(const float* __restrict__ s, _Float16* __restrict__ d,
                             size_t n) {
  for (size_t i = (size_t)blockIdx.x * blockDim.x + threadIdx.x; i < n;
       i += (size_t)gridDim.x * blockDim.x)
    d[i] = (_Float16)s[i];
}

// ---------------------------------------------------------------------------
// RMSNorm: one workgroup per token (2048 f32 in -> 2048 f16 out)
// ---------------------------------------------------------------------------
__global__ void __launch_bounds__(256)
rmsnorm_kernel(const float* __restrict__ x, const float* __restrict__ w,
               _Float16* __restrict__ out) {
  const int t = blockIdx.x;
  const int tid = threadIdx.x;
  const int lane = tid & 31, wid = tid >> 5;
  __shared__ float partial[8];
  const float* xr = x + (size_t)t * 2048;
  float ss = 0.f;
#pragma unroll
  for (int i = 0; i < 8; ++i) {
    float v = xr[tid + i * 256];
    ss += v * v;
  }
#pragma unroll
  for (int off = 1; off < 32; off <<= 1) ss += __shfl_xor(ss, off, 32);
  if (lane == 0) partial[wid] = ss;
  __syncthreads();
  if (tid == 0) {
    float tot = 0.f;
#pragma unroll
    for (int i = 0; i < 8; ++i) tot += partial[i];
    partial[0] = rsqrtf(tot * (1.0f / 2048.0f) + 1e-6f);
  }
  __syncthreads();
  float nrm = partial[0];
  _Float16* orow = out + (size_t)t * 2048;
#pragma unroll
  for (int i = 0; i < 8; ++i) {
    int idx = tid + i * 256;
    orow[idx] = (_Float16)(xr[idx] * nrm * w[idx]);
  }
}

// ---------------------------------------------------------------------------
// WMMA GEMM, double-buffered & software pipelined.
// C[M,N] = A[M,K](f16) * B[K,N](f16)  [+ resid(f32)]
// 128x128x32 tiles, 8 waves, each wave -> 64x32 (8 accumulating WMMAs/step).
// A tiles: global_load_async_to_lds_b128.  B tiles: VGPR prefetch one step
// ahead, transpose-store to LDS.  MODE 0: f16 out.  MODE 1: +resid, f32 out.
// ---------------------------------------------------------------------------
template <int MODE>
__global__ void __launch_bounds__(256)
gemm_wmma(const _Float16* __restrict__ A, const _Float16* __restrict__ Bw,
          const float* __restrict__ resid, void* __restrict__ outp,
          int M, int N, int K) {
  constexpr int LDT = 40;  // halves; 80B rows keep 16B chunk alignment
  __shared__ __align__(16) _Float16 Ash[2][128 * LDT];
  __shared__ __align__(16) _Float16 Bsh[2][128 * LDT];  // transposed [n][k]
  const int tid = threadIdx.x;
  const int lane = tid & 31, wid = tid >> 5;
  const int hi = lane >> 4, lo16 = lane & 15;
  const int waveM = wid & 1, waveN = wid >> 1;
  const int mBase = blockIdx.y * 128, nBase = blockIdx.x * 128;
  const int arow = tid >> 1, acc4 = (tid & 1) * 16;      // A chunk coords
  const int bk = tid >> 4, bn0 = (tid & 15) * 8;         // B chunk coords

  v8f acc[4][2];
#pragma unroll
  for (int i = 0; i < 4; ++i)
#pragma unroll
    for (int j = 0; j < 2; ++j)
#pragma unroll
      for (int r = 0; r < 8; ++r) acc[i][j][r] = 0.f;

  auto issueA = [&](int k0, int buf) {
    // 128x32 halves: each thread DMAs two 16B chunks, global->LDS direct.
    async_load_b128(&A[(size_t)(mBase + arow) * K + k0 + acc4],
                    &Ash[buf][arow * LDT + acc4]);
    async_load_b128(&A[(size_t)(mBase + arow) * K + k0 + acc4 + 8],
                    &Ash[buf][arow * LDT + acc4 + 8]);
  };
  auto loadB = [&](int k0, v8h bv[2]) {
#pragma unroll
    for (int p = 0; p < 2; ++p)
      bv[p] = *(const v8h*)&Bw[(size_t)(k0 + p * 16 + bk) * N + nBase + bn0];
  };
  auto storeB = [&](int buf, v8h bv[2]) {
#pragma unroll
    for (int p = 0; p < 2; ++p)
#pragma unroll
      for (int i = 0; i < 8; ++i)
        Bsh[buf][(bn0 + i) * LDT + p * 16 + bk] = bv[p][i];
  };

  v8h bv[2];
  issueA(0, 0);
  loadB(0, bv);
  const int nk = K >> 5;

  for (int kt = 0; kt < nk; ++kt) {
    const int cur = kt & 1, nxt = cur ^ 1;
    storeB(cur, bv);   // transpose B tile for this step into LDS
    wait_async0();     // this wave's A[cur] DMA complete
    __syncthreads();   // all waves' fills visible; prev reads of bufs done
    if (kt + 1 < nk) {
      issueA((kt + 1) << 5, nxt);  // DMA next A tile while computing
      loadB((kt + 1) << 5, bv);    // prefetch next B tile into VGPRs
    }

    v16h af[4];
#pragma unroll
    for (int fm = 0; fm < 4; ++fm) {
      const _Float16* ar = &Ash[cur][(waveM * 64 + fm * 16 + lo16) * LDT];
      af[fm] = cat8(*(const v8h*)&ar[hi * 8], *(const v8h*)&ar[16 + hi * 8]);
    }
    v16h bf[2];
#pragma unroll
    for (int fn = 0; fn < 2; ++fn) {
      const _Float16* br = &Bsh[cur][(waveN * 32 + fn * 16 + lo16) * LDT];
      bf[fn] = cat8(*(const v8h*)&br[hi * 8], *(const v8h*)&br[16 + hi * 8]);
    }
#pragma unroll
    for (int fm = 0; fm < 4; ++fm)
#pragma unroll
      for (int fn = 0; fn < 2; ++fn)
        acc[fm][fn] = wmma_f16(af[fm], bf[fn], acc[fm][fn]);
  }

#pragma unroll
  for (int fm = 0; fm < 4; ++fm)
#pragma unroll
    for (int fn = 0; fn < 2; ++fn)
#pragma unroll
      for (int r = 0; r < 8; ++r) {
        int grow = mBase + waveM * 64 + fm * 16 + hi * 8 + r;
        int gcol = nBase + waveN * 32 + fn * 16 + lo16;
        size_t idx = (size_t)grow * N + gcol;
        if (MODE == 0) {
          ((_Float16*)outp)[idx] = (_Float16)acc[fm][fn][r];
        } else {
          ((float*)outp)[idx] = resid[idx] + acc[fm][fn][r];
        }
      }
}

// ---------------------------------------------------------------------------
// Fused gate/up GEMM + SiLU*up -> f16. Same pipeline, two B streams.
// ---------------------------------------------------------------------------
__global__ void __launch_bounds__(256)
gateup_wmma(const _Float16* __restrict__ A, const _Float16* __restrict__ Gw,
            const _Float16* __restrict__ Uw, _Float16* __restrict__ outp,
            int M, int N, int K) {
  constexpr int LDT = 40;
  __shared__ __align__(16) _Float16 Ash[2][128 * LDT];
  __shared__ __align__(16) _Float16 Gsh[2][128 * LDT];
  __shared__ __align__(16) _Float16 Ush[2][128 * LDT];
  const int tid = threadIdx.x;
  const int lane = tid & 31, wid = tid >> 5;
  const int hi = lane >> 4, lo16 = lane & 15;
  const int waveM = wid & 1, waveN = wid >> 1;
  const int mBase = blockIdx.y * 128, nBase = blockIdx.x * 128;
  const int arow = tid >> 1, acc4 = (tid & 1) * 16;
  const int bk = tid >> 4, bn0 = (tid & 15) * 8;

  v8f accG[4][2], accU[4][2];
#pragma unroll
  for (int i = 0; i < 4; ++i)
#pragma unroll
    for (int j = 0; j < 2; ++j)
#pragma unroll
      for (int r = 0; r < 8; ++r) { accG[i][j][r] = 0.f; accU[i][j][r] = 0.f; }

  auto issueA = [&](int k0, int buf) {
    async_load_b128(&A[(size_t)(mBase + arow) * K + k0 + acc4],
                    &Ash[buf][arow * LDT + acc4]);
    async_load_b128(&A[(size_t)(mBase + arow) * K + k0 + acc4 + 8],
                    &Ash[buf][arow * LDT + acc4 + 8]);
  };
  auto loadGU = [&](int k0, v8h gv[2], v8h uv[2]) {
#pragma unroll
    for (int p = 0; p < 2; ++p) {
      size_t off = (size_t)(k0 + p * 16 + bk) * N + nBase + bn0;
      gv[p] = *(const v8h*)&Gw[off];
      uv[p] = *(const v8h*)&Uw[off];
    }
  };
  auto storeGU = [&](int buf, v8h gv[2], v8h uv[2]) {
#pragma unroll
    for (int p = 0; p < 2; ++p)
#pragma unroll
      for (int i = 0; i < 8; ++i) {
        Gsh[buf][(bn0 + i) * LDT + p * 16 + bk] = gv[p][i];
        Ush[buf][(bn0 + i) * LDT + p * 16 + bk] = uv[p][i];
      }
  };

  v8h gv[2], uv[2];
  issueA(0, 0);
  loadGU(0, gv, uv);
  const int nk = K >> 5;

  for (int kt = 0; kt < nk; ++kt) {
    const int cur = kt & 1, nxt = cur ^ 1;
    storeGU(cur, gv, uv);
    wait_async0();
    __syncthreads();
    if (kt + 1 < nk) {
      issueA((kt + 1) << 5, nxt);
      loadGU((kt + 1) << 5, gv, uv);
    }

    v16h af[4];
#pragma unroll
    for (int fm = 0; fm < 4; ++fm) {
      const _Float16* ar = &Ash[cur][(waveM * 64 + fm * 16 + lo16) * LDT];
      af[fm] = cat8(*(const v8h*)&ar[hi * 8], *(const v8h*)&ar[16 + hi * 8]);
    }
#pragma unroll
    for (int fn = 0; fn < 2; ++fn) {
      const _Float16* gr = &Gsh[cur][(waveN * 32 + fn * 16 + lo16) * LDT];
      const _Float16* ur = &Ush[cur][(waveN * 32 + fn * 16 + lo16) * LDT];
      v16h bg = cat8(*(const v8h*)&gr[hi * 8], *(const v8h*)&gr[16 + hi * 8]);
      v16h bu = cat8(*(const v8h*)&ur[hi * 8], *(const v8h*)&ur[16 + hi * 8]);
#pragma unroll
      for (int fm = 0; fm < 4; ++fm) {
        accG[fm][fn] = wmma_f16(af[fm], bg, accG[fm][fn]);
        accU[fm][fn] = wmma_f16(af[fm], bu, accU[fm][fn]);
      }
    }
  }

#pragma unroll
  for (int fm = 0; fm < 4; ++fm)
#pragma unroll
    for (int fn = 0; fn < 2; ++fn)
#pragma unroll
      for (int r = 0; r < 8; ++r) {
        int grow = mBase + waveM * 64 + fm * 16 + hi * 8 + r;
        int gcol = nBase + waveN * 32 + fn * 16 + lo16;
        float g = accG[fm][fn][r];
        float u = accU[fm][fn][r];
        // silu(g)*u with v_rcp_f32 instead of IEEE divide
        float h = g * __builtin_amdgcn_rcpf(1.f + __expf(-g)) * u;
        outp[(size_t)grow * N + gcol] = (_Float16)h;
      }
}

// ---------------------------------------------------------------------------
// Flash attention (causal). Block = 128 queries for one (batch, head).
// 8 waves; wave w owns 16 query rows. KV tiles of 32 keys through LDS.
// K tile via async-to-LDS; V transposed manually; QK^T and P*V on WMMA.
// ---------------------------------------------------------------------------
__global__ void __launch_bounds__(256)
attn_kernel(const _Float16* __restrict__ qkv, _Float16* __restrict__ out) {
  constexpr int LDK = 136;  // K tile row stride (halves)
  constexpr int LDV = 40;   // V^T / P row stride (halves)
  __shared__ __align__(16) _Float16 Ksh[32 * LDK];    // [key][dim]
  __shared__ __align__(16) _Float16 Vtsh[128 * LDV];  // [dim][key]
  __shared__ __align__(16) _Float16 Psh[8 * 16 * LDV];
  const int tid = threadIdx.x;
  const int lane = tid & 31, wid = tid >> 5;
  const int hi = lane >> 4, lo16 = lane & 15;
  const int qtile = blockIdx.x, head = blockIdx.y, bb = blockIdx.z;
  const int qBase = qtile * 128;
  const size_t tokBase = (size_t)bb * 2048;
  const float scale = 0.08838834764831845f;  // 1/sqrt(128)

  // Q fragments (A operand, 16x32 per frag, 4 frags cover head_dim=128)
  v16h qf[4];
  {
    size_t qrow = tokBase + qBase + wid * 16 + lo16;
    const _Float16* qp = &qkv[qrow * 6144 + (size_t)head * 128];
#pragma unroll
    for (int kf = 0; kf < 4; ++kf)
      qf[kf] = cat8(*(const v8h*)&qp[kf * 32 + hi * 8],
                    *(const v8h*)&qp[kf * 32 + 16 + hi * 8]);
  }

  float mrow[8], lrow[8];
  v8f o[8];
#pragma unroll
  for (int r = 0; r < 8; ++r) { mrow[r] = -3.0e38f; lrow[r] = 0.f; }
#pragma unroll
  for (int nf = 0; nf < 8; ++nf)
#pragma unroll
    for (int r = 0; r < 8; ++r) o[nf][r] = 0.f;

  _Float16* Pw = &Psh[wid * 16 * LDV];
  const int ntiles = (qBase + 128) / 32;

  for (int t = 0; t < ntiles; ++t) {
    const int kv0 = t * 32;
    __syncthreads();
    // K tile: 32 keys x 128 dims, row-major -> async DMA straight to LDS
#pragma unroll
    for (int p = 0; p < 2; ++p) {
      int c = p * 256 + tid;
      int row = c >> 4, cc = c & 15;
      async_load_b128(&qkv[(tokBase + kv0 + row) * 6144 + 2048 +
                           (size_t)head * 128 + cc * 8],
                      &Ksh[row * LDK + cc * 8]);
    }
    // V tile transposed: [dim][key]
#pragma unroll
    for (int p = 0; p < 2; ++p) {
      int k = p * 16 + (tid >> 4);
      int n0 = (tid & 15) * 8;
      v8h ch = *(const v8h*)&qkv[(tokBase + kv0 + k) * 6144 + 4096 +
                                 (size_t)head * 128 + n0];
#pragma unroll
      for (int i = 0; i < 8; ++i) Vtsh[(n0 + i) * LDV + k] = ch[i];
    }
    wait_async0();
    __syncthreads();

    // S = Q * K^T : 16 queries x 32 keys
    v8f sf[2];
#pragma unroll
    for (int f = 0; f < 2; ++f) {
      v8f a;
#pragma unroll
      for (int r = 0; r < 8; ++r) a[r] = 0.f;
#pragma unroll
      for (int kf = 0; kf < 4; ++kf) {
        const _Float16* kr = &Ksh[(f * 16 + lo16) * LDK + kf * 32];
        v16h bfr = cat8(*(const v8h*)&kr[hi * 8], *(const v8h*)&kr[16 + hi * 8]);
        a = wmma_f16(qf[kf], bfr, a);
      }
      sf[f] = a;
    }

    // Online softmax update (per lane: rows hi*8 .. hi*8+7)
#pragma unroll
    for (int r = 0; r < 8; ++r) {
      int qrow = qBase + wid * 16 + hi * 8 + r;
      float s0 = sf[0][r] * scale;
      float s1 = sf[1][r] * scale;
      if (kv0 + lo16 > qrow) s0 = -3.0e38f;        // causal mask
      if (kv0 + 16 + lo16 > qrow) s1 = -3.0e38f;
      float tm = halfmax16(fmaxf(s0, s1));
      float mnew = fmaxf(mrow[r], tm);
      float corr = __expf(mrow[r] - mnew);
      float p0 = __expf(s0 - mnew);
      float p1 = __expf(s1 - mnew);
      lrow[r] = lrow[r] * corr + halfsum16(p0 + p1);
      mrow[r] = mnew;
#pragma unroll
      for (int nf = 0; nf < 8; ++nf) o[nf][r] *= corr;
      int prow = hi * 8 + r;
      Pw[prow * LDV + lo16] = (_Float16)p0;
      Pw[prow * LDV + 16 + lo16] = (_Float16)p1;
    }
    // wave-local LDS RAW: wait for the P stores before re-reading as A-frag
    wait_ds0();

    const _Float16* pr = &Pw[lo16 * LDV];
    v16h pa = cat8(*(const v8h*)&pr[hi * 8], *(const v8h*)&pr[16 + hi * 8]);
#pragma unroll
    for (int nf = 0; nf < 8; ++nf) {
      const _Float16* vr = &Vtsh[(nf * 16 + lo16) * LDV];
      v16h bv = cat8(*(const v8h*)&vr[hi * 8], *(const v8h*)&vr[16 + hi * 8]);
      o[nf] = wmma_f16(pa, bv, o[nf]);
    }
  }

  // epilogue: O / l -> f16, heads concatenated along hidden dim
#pragma unroll
  for (int r = 0; r < 8; ++r) {
    float inv = __builtin_amdgcn_rcpf(lrow[r]);
    size_t orow = tokBase + qBase + wid * 16 + hi * 8 + r;
    _Float16* op = &out[orow * 2048 + (size_t)head * 128];
#pragma unroll
    for (int nf = 0; nf < 8; ++nf)
      op[nf * 16 + lo16] = (_Float16)(o[nf][r] * inv);
  }
}

// ---------------------------------------------------------------------------
extern "C" void kernel_launch(void* const* d_in, const int* in_sizes, int n_in,
                              void* d_out, int out_size, void* d_ws, size_t ws_size,
                              hipStream_t stream) {
  (void)in_sizes; (void)n_in; (void)out_size; (void)ws_size;
  const float* x       = (const float*)d_in[0];
  const float* attn_nw = (const float*)d_in[1];
  const float* ff_nw   = (const float*)d_in[2];
  const float* qkv_w   = (const float*)d_in[3];
  const float* out_w   = (const float*)d_in[4];
  const float* gate_w  = (const float*)d_in[5];
  const float* up_w    = (const float*)d_in[6];
  const float* down_w  = (const float*)d_in[7];
  float* out = (float*)d_out;

  const size_t H = 2048, I = 8192, TOK = 4096;
  char* p = (char*)d_ws;
  auto alloc = [&](size_t bytes) {
    char* r = p;
    p += (bytes + 255) & ~(size_t)255;
    return r;
  };
  _Float16* qkv_w16  = (_Float16*)alloc(H * 3 * H * 2);
  _Float16* out_w16  = (_Float16*)alloc(H * H * 2);
  _Float16* gate_w16 = (_Float16*)alloc(H * I * 2);
  _Float16* up_w16   = (_Float16*)alloc(H * I * 2);
  _Float16* down_w16 = (_Float16*)alloc(I * H * 2);
  _Float16* xn16     = (_Float16*)alloc(TOK * H * 2);
  _Float16* qkv16    = (_Float16*)alloc(TOK * 3 * H * 2);
  _Float16* attn16   = (_Float16*)alloc(TOK * H * 2);
  float*    x1       = (float*)alloc(TOK * H * 4);
  _Float16* x1n16    = (_Float16*)alloc(TOK * H * 2);
  _Float16* h16      = (_Float16*)alloc(TOK * I * 2);

  // 1) weights -> f16 (single HBM pass; re-reads hit 192MB L2)
  conv_f32_f16<<<2048, 256, 0, stream>>>(qkv_w,  qkv_w16,  H * 3 * H);
  conv_f32_f16<<<2048, 256, 0, stream>>>(out_w,  out_w16,  H * H);
  conv_f32_f16<<<2048, 256, 0, stream>>>(gate_w, gate_w16, H * I);
  conv_f32_f16<<<2048, 256, 0, stream>>>(up_w,   up_w16,   H * I);
  conv_f32_f16<<<2048, 256, 0, stream>>>(down_w, down_w16, I * H);

  // 2) attn RMSNorm
  rmsnorm_kernel<<<4096, 256, 0, stream>>>(x, attn_nw, xn16);
  // 3) QKV GEMM (4096 x 6144 x 2048) -> f16
  gemm_wmma<0><<<dim3(48, 32), 256, 0, stream>>>(xn16, qkv_w16, nullptr,
                                                 (void*)qkv16, 4096, 6144, 2048);
  // 4) causal flash attention -> f16
  attn_kernel<<<dim3(16, 16, 2), 256, 0, stream>>>(qkv16, attn16);
  // 5) out-proj + residual -> x1 (f32)
  gemm_wmma<1><<<dim3(16, 32), 256, 0, stream>>>(attn16, out_w16, x,
                                                 (void*)x1, 4096, 2048, 2048);
  // 6) ffn RMSNorm
  rmsnorm_kernel<<<4096, 256, 0, stream>>>(x1, ff_nw, x1n16);
  // 7) fused gate/up GEMM + SiLU*up (4096 x 8192 x 2048) -> f16
  gateup_wmma<<<dim3(64, 32), 256, 0, stream>>>(x1n16, gate_w16, up_w16, h16,
                                                4096, 8192, 2048);
  // 8) down GEMM + residual -> output (f32)
  gemm_wmma<1><<<dim3(16, 32), 256, 0, stream>>>(h16, down_w16, x1,
                                                 (void*)out, 4096, 2048, 8192);
}
